// GRUClassifier_5360119185988
// MI455X (gfx1250) — compile-verified, matrix operand order checked by reading, not run on hardware
//
#include <hip/hip_runtime.h>
#include <hip/hip_bf16.h>
#include <hip/hip_fp16.h>

typedef _Float16 half16 __attribute__((ext_vector_type(16)));
typedef _Float16 half2v __attribute__((ext_vector_type(2)));
typedef float    float8 __attribute__((ext_vector_type(8)));

#define T_  512
#define B_  4096
#define IN_ 12
#define H_  32
#define L_  10
#define C_  4
#define G3  96

// ws layout (bytes): [WI f16 10*96*32 | WH f16 10*96*32 | BRZ f32 10*64 | BIN f32 10*32 | BHN f32 10*32]
#define WS_WI  0
#define WS_WH  61440
#define WS_BRZ 122880
#define WS_BIN 125440
#define WS_BHN 126720
#define WS_WB_BYTES 128000

#define NWAVES 2                    // waves per block
#define LDS_SLOTS  128000           // A-layout h slot region offset in LDS
#define SLOT_HALVES 512             // 16 rows * 32 halves per layer slot
#define WAVE_SLOT_BYTES (L_ * 1024)
#define LDS_HS (LDS_SLOTS + NWAVES * WAVE_SLOT_BYTES)          // 148480: per-lane packed h state
#define HS_BYTES (NWAVES * 32 * L_ * 32)                       // 2*32*10*8 u32 = 20480 B
#define LDS_TOTAL (LDS_HS + HS_BYTES)                          // 168960 B

// position of K index k inside a 32-half A/B lane record:
// halves [0..15] -> K {0..7,16..23} (lanes 0-15), halves [16..31] -> K {8..15,24..31} (lanes 16-31)
__device__ __host__ __forceinline__ int kperm(int k) {
  return (k < 8) ? k : (k < 16) ? (k + 8) : (k < 24) ? (k - 8) : k;
}

__device__ __forceinline__ float8 wmma_f16(half16 a, half16 b, float8 c) {
  return __builtin_amdgcn_wmma_f32_16x16x32_f16(false, a, false, b, (short)0, c, false, false);
}

__device__ __forceinline__ float sigmoid_f(float x) { return 1.f / (1.f + __expf(-x)); }
__device__ __forceinline__ float tanh_f(float x)    { return 2.f / (1.f + __expf(-2.f * x)) - 1.f; }

// ---------------- weight prep: transpose + K-permute + f16 convert ----------------
__global__ void gru_prep_kernel(const float* __restrict__ W_ih0, const float* __restrict__ W_hh0,
                                const float* __restrict__ b_ih0, const float* __restrict__ b_hh0,
                                const float* __restrict__ W_ih,  const float* __restrict__ W_hh,
                                const float* __restrict__ b_ih,  const float* __restrict__ b_hh,
                                char* __restrict__ ws) {
  int idx = blockIdx.x * 256 + threadIdx.x;
  _Float16* wi  = (_Float16*)(ws + WS_WI);
  _Float16* wh  = (_Float16*)(ws + WS_WH);
  float*    brz = (float*)(ws + WS_BRZ);
  float*    bin = (float*)(ws + WS_BIN);
  float*    bhn = (float*)(ws + WS_BHN);
  if (idx < L_ * G3 * 32) {
    int k = idx & 31, n = (idx >> 5) % G3, l = idx / (32 * G3);
    float vi, vh;
    if (l == 0) {
      vi = (k < IN_) ? W_ih0[n * IN_ + k] : 0.f;   // zero-pad K 12..31
      vh = W_hh0[n * H_ + k];
    } else {
      vi = W_ih[((l - 1) * G3 + n) * H_ + k];
      vh = W_hh[((l - 1) * G3 + n) * H_ + k];
    }
    int p = (l * G3 + n) * 32 + kperm(k);
    wi[p] = (_Float16)vi;
    wh[p] = (_Float16)vh;
  }
  if (idx < L_ * 128) {
    int j = idx & 127, l = idx >> 7;
    const float* bi = (l == 0) ? b_ih0 : (b_ih + (l - 1) * G3);
    const float* bh = (l == 0) ? b_hh0 : (b_hh + (l - 1) * G3);
    if (j < 64)      brz[l * 64 + j]        = bi[j] + bh[j];   // r,z: combined bias
    else if (j < 96) bin[l * 32 + (j - 64)] = bi[j];           // n: input-path bias
    else             bhn[l * 32 + (j - 96)] = bh[j];           // n: hidden-path bias
  }
}

// ---------------- persistent GRU kernel: 1 wave = 16 batch rows, all T, all L ----------------
__global__ __launch_bounds__(32 * NWAVES, 1) void gru_main_kernel(
    const float* __restrict__ x, const char* __restrict__ ws,
    const float* __restrict__ fc_W, const float* __restrict__ fc_b,
    const int* __restrict__ lengths, float* __restrict__ out) {
  extern __shared__ char lds[];
  const int tid = threadIdx.x;

  // cooperative copy of weights/biases ws -> LDS, zero the h slots + h state
  {
    const uint4* src = (const uint4*)ws;
    uint4* dst = (uint4*)lds;
    for (int i = tid; i < WS_WB_BYTES / 16; i += 32 * NWAVES) dst[i] = src[i];
    uint4 z; z.x = 0; z.y = 0; z.z = 0; z.w = 0;
    uint4* sl = (uint4*)(lds + LDS_SLOTS);
    for (int i = tid; i < (LDS_TOTAL - LDS_SLOTS) / 16; i += 32 * NWAVES) sl[i] = z;
  }
  __syncthreads();

  const int w     = tid >> 5;
  const int lane  = tid & 31;
  const int half  = lane >> 4;
  const int col16 = lane & 15;
  const int b0    = (blockIdx.x * NWAVES + w) * 16;

  const _Float16* wiL  = (const _Float16*)(lds + WS_WI);
  const _Float16* whL  = (const _Float16*)(lds + WS_WH);
  const float*    brzL = (const float*)(lds + WS_BRZ);
  const float*    binL = (const float*)(lds + WS_BIN);
  const float*    bhnL = (const float*)(lds + WS_BHN);
  _Float16* slot = (_Float16*)(lds + LDS_SLOTS + w * WAVE_SLOT_BYTES);  // [L_][512] halves, A-layout h
  // per-lane private packed h state: [L_][8] u32 per lane (element (r+half*8, jt*16+col16) in half jt)
  unsigned* hsLane = (unsigned*)(lds + LDS_HS) + (size_t)(w * 32 + lane) * L_ * 8;

  int lenr[8];
#pragma unroll
  for (int r = 0; r < 8; ++r) lenr[r] = lengths[b0 + r + half * 8];

  unsigned lastp[8];
#pragma unroll
  for (int r = 0; r < 8; ++r) lastp[r] = 0u;

  const float8 ZACC = {};                        // inline-0 C operand for first WMMA of each chain
  const int posBase = col16 + (col16 & 8);       // kperm(c) = posBase + jt*8 for c = jt*16+col16
  const int aOff    = col16 * 32 + half * 16;    // per-lane A-operand record offset (halves)

  for (int t = 0; t < T_; ++t) {
    // ---- build layer-0 A operand from global x (K 12..31 zero-padded) ----
    half16 ax;
    {
      const float* xp = x + ((size_t)t * B_ + b0 + col16) * IN_;
      float f[8];
      if (half == 0) {
        float4 a = *(const float4*)xp;
        float4 b = *(const float4*)(xp + 4);
        f[0] = a.x; f[1] = a.y; f[2] = a.z; f[3] = a.w;
        f[4] = b.x; f[5] = b.y; f[6] = b.z; f[7] = b.w;
      } else {
        float4 a = *(const float4*)(xp + 8);
        f[0] = a.x; f[1] = a.y; f[2] = a.z; f[3] = a.w;
        f[4] = 0.f; f[5] = 0.f; f[6] = 0.f; f[7] = 0.f;
      }
#pragma unroll
      for (int i = 0; i < 8; ++i) ax[i] = (_Float16)f[i];
#pragma unroll
      for (int i = 8; i < 16; ++i) ax[i] = (_Float16)0.f;
    }

#pragma unroll
    for (int l = 0; l < L_; ++l) {
      // ---- issue all DS loads up front with distinct destinations (partial dscnt waits) ----
      if (l > 0) ax = *(const half16*)(slot + (l - 1) * SLOT_HALVES + aOff);  // prev layer's output
      half16 ah = *(const half16*)(slot + l * SLOT_HALVES + aOff);            // this layer's h(t-1)

      const _Float16* wiB = wiL + l * G3 * 32;
      const _Float16* whB = whL + l * G3 * 32;
      half16 wti[6], wth[6];
#pragma unroll
      for (int j = 0; j < 6; ++j) wti[j] = *(const half16*)(wiB + j * 512 + aOff);
#pragma unroll
      for (int j = 0; j < 6; ++j) wth[j] = *(const half16*)(whB + j * 512 + aOff);

      // per-lane bias scalars (depend on col16 only)
      float bR0 = brzL[l * 64 + col16],      bR1 = brzL[l * 64 + 16 + col16];
      float bZ0 = brzL[l * 64 + 32 + col16], bZ1 = brzL[l * 64 + 48 + col16];
      float bI0 = binL[l * 32 + col16],      bI1 = binL[l * 32 + 16 + col16];
      float bH0 = bhnL[l * 32 + col16],      bH1 = bhnL[l * 32 + 16 + col16];

      // packed h(t-1) state for this layer (per-lane private)
      uint4 ho0 = *(const uint4*)(hsLane + l * 8);
      uint4 ho1 = *(const uint4*)(hsLane + l * 8 + 4);

      // ---- 12 WMMAs: gi = x@Wi^T (C = inline 0) then gh = h@Wh^T chained; n-gate split ----
      float8 arz0 = wmma_f16(ax, wti[0], ZACC);
      float8 arz1 = wmma_f16(ax, wti[1], ZACC);
      float8 arz2 = wmma_f16(ax, wti[2], ZACC);
      float8 arz3 = wmma_f16(ax, wti[3], ZACC);
      float8 ain0 = wmma_f16(ax, wti[4], ZACC);
      float8 ain1 = wmma_f16(ax, wti[5], ZACC);
      arz0 = wmma_f16(ah, wth[0], arz0);
      arz1 = wmma_f16(ah, wth[1], arz1);
      arz2 = wmma_f16(ah, wth[2], arz2);
      arz3 = wmma_f16(ah, wth[3], arz3);
      float8 ahn0 = wmma_f16(ah, wth[4], ZACC);
      float8 ahn1 = wmma_f16(ah, wth[5], ZACC);

      // ---- lane-local GRU gate math; state as packed f16 pairs (jt=0 low, jt=1 high) ----
      unsigned hw[8], nw[8];
      hw[0] = ho0.x; hw[1] = ho0.y; hw[2] = ho0.z; hw[3] = ho0.w;
      hw[4] = ho1.x; hw[5] = ho1.y; hw[6] = ho1.z; hw[7] = ho1.w;
#pragma unroll
      for (int r = 0; r < 8; ++r) {
        half2v hold = __builtin_bit_cast(half2v, hw[r]);
        half2v hnew;
#pragma unroll
        for (int jt = 0; jt < 2; ++jt) {
          float g_r  = (jt ? arz1[r] : arz0[r]) + (jt ? bR1 : bR0);
          float g_z  = (jt ? arz3[r] : arz2[r]) + (jt ? bZ1 : bZ0);
          float g_in = (jt ? ain1[r] : ain0[r]) + (jt ? bI1 : bI0);
          float g_hn = (jt ? ahn1[r] : ahn0[r]) + (jt ? bH1 : bH0);
          float rg = sigmoid_f(g_r);
          float zg = sigmoid_f(g_z);
          float ng = tanh_f(g_in + rg * g_hn);
          float hv = (1.f - zg) * ng + zg * (float)hold[jt];
          hnew[jt] = (_Float16)hv;
          // transpose-store into A-layout f16 slot for next layer / next timestep
          slot[l * SLOT_HALVES + (r + half * 8) * 32 + posBase + jt * 8] = hnew[jt];
        }
        nw[r] = __builtin_bit_cast(unsigned, hnew);
        if (l == L_ - 1) {
          if (t == lenr[r] - 1) lastp[r] = nw[r];  // capture h at last valid step
        }
      }
      uint4 hn0, hn1;
      hn0.x = nw[0]; hn0.y = nw[1]; hn0.z = nw[2]; hn0.w = nw[3];
      hn1.x = nw[4]; hn1.y = nw[5]; hn1.z = nw[6]; hn1.w = nw[7];
      *(uint4*)(hsLane + l * 8)     = hn0;
      *(uint4*)(hsLane + l * 8 + 4) = hn1;
      __builtin_amdgcn_wave_barrier();  // keep ds stores before next layer's ds loads (HW is in-order)
    }
  }

  // ---- epilogue: FC (32 -> 4) + log_softmax ----
  float* lb = (float*)(lds + LDS_SLOTS + w * WAVE_SLOT_BYTES);  // reuse wave slot as [16][32] f32
#pragma unroll
  for (int r = 0; r < 8; ++r) {
    half2v lp = __builtin_bit_cast(half2v, lastp[r]);
#pragma unroll
    for (int jt = 0; jt < 2; ++jt)
      lb[(r + half * 8) * 32 + jt * 16 + col16] = (float)lp[jt];
  }
  __builtin_amdgcn_wave_barrier();

  if (lane < 16) {
    int row = lane;
    float lg[4];
#pragma unroll
    for (int c = 0; c < 4; ++c) {
      float s = fc_b[c];
#pragma unroll
      for (int j = 0; j < 32; ++j) s += lb[row * 32 + j] * fc_W[c * 32 + j];
      lg[c] = s;
    }
    float mx = fmaxf(fmaxf(lg[0], lg[1]), fmaxf(lg[2], lg[3]));
    float se = 0.f;
#pragma unroll
    for (int c = 0; c < 4; ++c) se += __expf(lg[c] - mx);
    float lse = mx + __logf(se);
#pragma unroll
    for (int c = 0; c < 4; ++c) out[(size_t)(b0 + row) * 4 + c] = lg[c] - lse;
  }
}

extern "C" void kernel_launch(void* const* d_in, const int* in_sizes, int n_in,
                              void* d_out, int out_size, void* d_ws, size_t ws_size,
                              hipStream_t stream) {
  const float* x     = (const float*)d_in[0];
  const float* W_ih0 = (const float*)d_in[1];
  const float* W_hh0 = (const float*)d_in[2];
  const float* b_ih0 = (const float*)d_in[3];
  const float* b_hh0 = (const float*)d_in[4];
  const float* W_ih  = (const float*)d_in[5];
  const float* W_hh  = (const float*)d_in[6];
  const float* b_ih  = (const float*)d_in[7];
  const float* b_hh  = (const float*)d_in[8];
  const float* fc_W  = (const float*)d_in[9];
  const float* fc_b  = (const float*)d_in[10];
  const int*   lens  = (const int*)d_in[11];
  (void)in_sizes; (void)n_in; (void)out_size; (void)ws_size;

  // weight prep: 10*96*32 = 30720 work items
  gru_prep_kernel<<<(L_ * G3 * 32 + 255) / 256, 256, 0, stream>>>(
      W_ih0, W_hh0, b_ih0, b_hh0, W_ih, W_hh, b_ih, b_hh, (char*)d_ws);

  // allow >64KB dynamic LDS (168,960 B per block; 320KB/WGP available on CDNA5)
  (void)hipFuncSetAttribute((const void*)gru_main_kernel,
                            hipFuncAttributeMaxDynamicSharedMemorySize, LDS_TOTAL);

  // 128 blocks * 2 waves * 16 rows = 4096 batch rows, spread across many WGPs
  gru_main_kernel<<<B_ / (16 * NWAVES), 32 * NWAVES, LDS_TOTAL, stream>>>(
      x, (const char*)d_ws, fc_W, fc_b, lens, (float*)d_out);
}
